// DocumentHierachicalInnerAttention_82772609728773
// MI455X (gfx1250) — matrix operand order repaired and checked by reading, not compiled, not run on hardware
//
#include <hip/hip_runtime.h>
#include <math.h>

#define B_  16
#define S_  32
#define LD_ 128
#define LQ_ 32
#define H_  256
#define A_  256
#define FH_ 1024   // 4*H

typedef __attribute__((ext_vector_type(16))) __bf16 v16bf;
typedef __attribute__((ext_vector_type(8)))  __bf16 v8bf;
typedef __attribute__((ext_vector_type(8)))  float  v8f;
typedef __attribute__((ext_vector_type(4)))  float  f4;

// ---------------------------------------------------------------------------
// WMMA wrapper: D = A(16x32 bf16) * B(32x16 bf16) + C(16x16 f32), wave32.
// ---------------------------------------------------------------------------
__device__ __forceinline__ v8f wmma_bf16(v16bf a, v16bf b, v8f c) {
  return __builtin_amdgcn_wmma_f32_16x16x32_bf16(
      /*neg_a=*/false, a, /*neg_b=*/false, b,
      /*c_mod=*/(short)0, c, /*reuse_a=*/false, /*reuse_b=*/false);
}

__device__ __forceinline__ void lane_mh(int& m, int& hx) {
  int lane = threadIdx.x & 31;
  m = lane & 15;     // row (A) / col (B,C) within tile
  hx = lane >> 4;    // half selector
}

__device__ __forceinline__ v16bf concat8(v8bf lo, v8bf hi) {
  return __builtin_shufflevector(lo, hi, 0, 1, 2, 3, 4, 5, 6, 7,
                                 8, 9, 10, 11, 12, 13, 14, 15);
}

__device__ __forceinline__ __bf16 tobf(float x) { return (__bf16)x; }

// A fragment (16x32 bf16) from row-major bf16 LDS [row][k] (16B aligned).
// Lane layout: e<8 -> K = hx*8 + e ; e>=8 -> K = 16 + hx*8 + (e-8).
__device__ __forceinline__ v16bf load_A_lds(const __bf16* base, int stride,
                                            int row0, int k0) {
  int m, hx; lane_mh(m, hx);
  const v8bf* p = (const v8bf*)(base + (row0 + m) * stride + k0 + hx * 8);
  return concat8(p[0], p[2]);    // elements [0..7] and [16..23]
}

// A fragment from row-major fp32 global (16B aligned), converted to bf16.
__device__ __forceinline__ v16bf load_A_g(const float* base, int stride,
                                          int row0, int k0) {
  int m, hx; lane_mh(m, hx);
  const f4* p = (const f4*)(base + (size_t)(row0 + m) * stride + k0 + hx * 8);
  f4 a0 = p[0], a1 = p[1], a2 = p[4], a3 = p[5];
  v16bf a;
#pragma unroll
  for (int e = 0; e < 4; ++e) {
    a[e]      = tobf(a0[e]);
    a[4 + e]  = tobf(a1[e]);
    a[8 + e]  = tobf(a2[e]);
    a[12 + e] = tobf(a3[e]);
  }
  return a;
}

// A fragment = (global fp32 row) * (broadcast weight vector w[k]).
__device__ __forceinline__ v16bf load_A_g_scaled(const float* base, int stride,
                                                 int row0, int k0,
                                                 const float* wv) {
  int m, hx; lane_mh(m, hx);
  const f4* p = (const f4*)(base + (size_t)(row0 + m) * stride + k0 + hx * 8);
  const f4* w = (const f4*)(wv + k0 + hx * 8);
  f4 a0 = p[0], a1 = p[1], a2 = p[4], a3 = p[5];
  f4 w0 = w[0], w1 = w[1], w2 = w[4], w3 = w[5];
  v16bf a;
#pragma unroll
  for (int e = 0; e < 4; ++e) {
    a[e]      = tobf(a0[e] * w0[e]);
    a[4 + e]  = tobf(a1[e] * w1[e]);
    a[8 + e]  = tobf(a2[e] * w2[e]);
    a[12 + e] = tobf(a3[e] * w3[e]);
  }
  return a;
}

// A fragment = elementwise product of global fp32 row and LDS bf16 row.
__device__ __forceinline__ v16bf load_A_prod(const float* g, int gstride, int grow0,
                                             const __bf16* sb, int sstride, int srow0,
                                             int k0) {
  int m, hx; lane_mh(m, hx);
  const f4*   p = (const f4*)(g + (size_t)(grow0 + m) * gstride + k0 + hx * 8);
  const v8bf* q = (const v8bf*)(sb + (srow0 + m) * sstride + k0 + hx * 8);
  f4 a0 = p[0], a1 = p[1], a2 = p[4], a3 = p[5];
  v8bf q0 = q[0], q1 = q[2];
  v16bf a;
#pragma unroll
  for (int e = 0; e < 4; ++e) {
    a[e]      = tobf(a0[e] * (float)q0[e]);
    a[4 + e]  = tobf(a1[e] * (float)q0[4 + e]);
    a[8 + e]  = tobf(a2[e] * (float)q1[e]);
    a[12 + e] = tobf(a3[e] * (float)q1[4 + e]);
  }
  return a;
}

// B fragment (32x16 bf16) from LDS stored as [N][K] row-major (16B aligned).
// Lane layout: e<8 -> K = hx*8 + e ; e>=8 -> K = 16 + hx*8 + (e-8), N = lane&15.
__device__ __forceinline__ v16bf load_B_lds(const __bf16* base, int stride,
                                            int n0, int k0) {
  int n, hx; lane_mh(n, hx);
  const v8bf* p = (const v8bf*)(base + (n0 + n) * stride + k0 + hx * 8);
  return concat8(p[0], p[2]);
}

// B fragment from global fp32 stored as [N][K] row-major (weight matrices).
__device__ __forceinline__ v16bf load_B_g(const float* base, int stride,
                                          int n0, int k0) {
  int n, hx; lane_mh(n, hx);
  const f4* p = (const f4*)(base + (size_t)(n0 + n) * stride + k0 + hx * 8);
  f4 b0 = p[0], b1 = p[1], b2 = p[4], b3 = p[5];
  v16bf b;
#pragma unroll
  for (int e = 0; e < 4; ++e) {
    b[e]      = tobf(b0[e]);
    b[4 + e]  = tobf(b1[e]);
    b[8 + e]  = tobf(b2[e]);
    b[12 + e] = tobf(b3[e]);
  }
  return b;
}

// B fragment from global fp32 stored as [K][N] (transpose access, strided).
__device__ __forceinline__ v16bf load_B_g_kmajor(const float* base, int rowstride,
                                                 int k0, int n0) {
  int n, hx; lane_mh(n, hx);
  const float* p = base + (size_t)(k0 + hx * 8) * rowstride + n0 + n;
  v16bf b;
#pragma unroll
  for (int e = 0; e < 16; ++e) b[e] = tobf(p[(size_t)(e < 8 ? e : e + 8) * rowstride]);
  return b;
}

// ---------------------------------------------------------------------------
// LDS layout (dynamic, ~130 KB; CDNA5 allows up to 320 KB per workgroup)
// ---------------------------------------------------------------------------
#define OFF_AD2Q   0         // bf16 [64][264]  = 33792 B (ALIAS: fp32 Uq stage 32768 B)
#define OFF_AQ2D   33792     // bf16 [64][264]  = 33792 B
#define OFF_SD2Q   67584     // bf16 [128][40]  = 10240 B
#define OFF_SQ2DT  77824     // bf16 [32][136]  =  8704 B
#define OFF_UQT    86528     // bf16 [256][40]  = 20480 B  (alias: sUq [32][264])
#define OFF_TT     107008    // bf16 [256][40]  = 20480 B  (alias: sSc [128][40])
#define OFF_D1     127488    // f32 [128]
#define OFF_D2     128000    // f32 [32]
#define OFF_SLOG   128128    // f32 [64]
#define OFF_ACC    128384    // f32 [1024]
#define OFF_STATS  132480    // f32 [4] : m_run, s_run, scale
#define SMEM_BYTES 132608

// ---------------------------------------------------------------------------
// Kernel 1: per (b,s) pair -> alpha[bs][1024]
// ---------------------------------------------------------------------------
__global__ __launch_bounds__(256)
void dhia_inner_kernel(const float* __restrict__ U_d, const float* __restrict__ U_q,
                       const int* __restrict__ d_mask, const int* __restrict__ q_mask,
                       const float* __restrict__ wc_w, const float* __restrict__ wc_b,
                       const float* __restrict__ Ws_w, const float* __restrict__ Ws_b,
                       const float* __restrict__ wcs_w, const float* __restrict__ wcs_b,
                       float* __restrict__ alpha_out) {
  extern __shared__ char smem[];
  __bf16* sAd2q  = (__bf16*)(smem + OFF_AD2Q);
  __bf16* sAq2d  = (__bf16*)(smem + OFF_AQ2D);
  __bf16* sSd2q  = (__bf16*)(smem + OFF_SD2Q);
  __bf16* sSq2dT = (__bf16*)(smem + OFF_SQ2DT);
  __bf16* sUqT   = (__bf16*)(smem + OFF_UQT);
  __bf16* sTT    = (__bf16*)(smem + OFF_TT);
  __bf16* sUq    = (__bf16*)(smem + OFF_UQT);   // alias, live only in P1
  __bf16* sSc    = (__bf16*)(smem + OFF_TT);    // alias, live only in P1/P2
  float*  sUqStage = (float*)(smem + OFF_AD2Q); // fp32 Uq [32][256], live pre-P3
  float*  d1     = (float*)(smem + OFF_D1);
  float*  d2     = (float*)(smem + OFF_D2);
  float*  slog   = (float*)(smem + OFF_SLOG);
  float*  acc    = (float*)(smem + OFF_ACC);
  float*  stats  = (float*)(smem + OFF_STATS);

  const int tid  = threadIdx.x;
  const int wave = tid >> 5;
  const int bs   = blockIdx.x;
  const int b    = bs >> 5;                       // S_ == 32
  const float* Ud = U_d + (size_t)bs * LD_ * H_;
  const float* Uq = U_q + (size_t)b * LQ_ * H_;
  const int* dm   = d_mask + bs * LD_;
  const int* qm   = q_mask + b * LQ_;
  const float* w1 = wc_w;
  const float* w3 = wc_w + 2 * H_;

  __builtin_prefetch(Ud, 0, 0);   // global_prefetch: warm this pair's Ud tile

  // ---- async-DMA Uq (32 KB fp32) into LDS stage (ASYNCcnt path) ----
  {
#pragma unroll
    for (int it = 0; it < 8; ++it) {
      int goff = (tid + it * 256) * 16;                 // byte offset, 16B/lane
      unsigned lds_off = (unsigned)(OFF_AD2Q + goff);   // dyn-LDS starts at 0
      asm volatile("global_load_async_to_lds_b128 %0, %1, %2"
                   :
                   : "v"(lds_off), "v"(goff), "s"(Uq)
                   : "memory");
    }
  }

  // ---- overlap: init accumulators + d1[i] = Ud[i]·w1 from global ----
  for (int c = tid; c < FH_; c += 256) acc[c] = 0.f;
  if (tid == 0) { stats[0] = -3.0e38f; stats[1] = 0.f; stats[2] = 0.f; }
  if (tid < LD_) {
    float s = 0.f;
    for (int k = 0; k < H_; ++k) s += Ud[(size_t)tid * H_ + k] * w1[k];
    d1[tid] = s;
  }
  asm volatile("s_wait_asynccnt 0" ::: "memory");
  __syncthreads();

  // ---- build sUq (bf16 [N=j][K=k]) and d2 from the LDS stage ----
  for (int idx = tid; idx < LQ_ * H_; idx += 256) {
    int j = idx >> 8, k = idx & 255;
    sUq[j * 264 + k] = (__bf16)sUqStage[idx];
  }
  if (tid >= 192 && tid < 192 + LQ_) {
    int j = tid - 192;
    const float* w2 = wc_w + H_;
    float s = 0.f;
    for (int k = 0; k < H_; ++k) s += sUqStage[j * H_ + k] * w2[k];
    d2[j] = s;
  }
  __syncthreads();

  // ---- P1: Sc = (Ud⊙w3) @ Uq^T + d1 + d2 + bias  -> sSc bf16 [128][40] ----
  {
    const float bias = wc_b[0];
    const int mt = wave;                       // 8 waves x 16 rows = 128
    for (int nt = 0; nt < 2; ++nt) {
      v8f cacc = {};
      for (int kk = 0; kk < H_; kk += 32) {
        v16bf a  = load_A_g_scaled(Ud, H_, mt * 16, kk, w3);
        v16bf bb = load_B_lds(sUq, 264, nt * 16, kk);
        cacc = wmma_bf16(a, bb, cacc);
      }
      int n, hx; lane_mh(n, hx);
      int col = nt * 16 + n;
#pragma unroll
      for (int r = 0; r < 8; ++r) {
        int row = mt * 16 + hx * 8 + r;
        sSc[row * 40 + col] = (__bf16)(cacc[r] + d1[row] + d2[col] + bias);
      }
    }
  }
  __syncthreads();

  // ---- P2: masked softmaxes (row -> sSd2q, col -> sSq2dT) ----
  if (tid < LD_) {
    const int i = tid;
    const int di = dm[i];
    float mx = -3.0e38f;
    for (int j = 0; j < LQ_; ++j) {
      bool ok = (di > 0) && (qm[j] > 0);
      float v = ok ? (float)sSc[i * 40 + j] : -1.0e9f;
      mx = fmaxf(mx, v);
    }
    float ssum = 0.f;
    for (int j = 0; j < LQ_; ++j) {
      bool ok = (di > 0) && (qm[j] > 0);
      float v = ok ? (float)sSc[i * 40 + j] : -1.0e9f;
      ssum += __expf(v - mx);
    }
    float inv = 1.f / ssum;
    for (int j = 0; j < LQ_; ++j) {
      bool ok = (di > 0) && (qm[j] > 0);
      float v = ok ? (float)sSc[i * 40 + j] : -1.0e9f;
      sSd2q[i * 40 + j] = (__bf16)(ok ? __expf(v - mx) * inv : 0.f);
    }
  }
  __syncthreads();
  if (tid < LQ_) {
    const int j = tid;
    const int qj = qm[j];
    float mx = -3.0e38f;
    for (int i = 0; i < LD_; ++i) {
      bool ok = (qj > 0) && (dm[i] > 0);
      float v = ok ? (float)sSc[i * 40 + j] : -1.0e9f;
      mx = fmaxf(mx, v);
    }
    float ssum = 0.f;
    for (int i = 0; i < LD_; ++i) {
      bool ok = (qj > 0) && (dm[i] > 0);
      float v = ok ? (float)sSc[i * 40 + j] : -1.0e9f;
      ssum += __expf(v - mx);
    }
    float inv = 1.f / ssum;
    for (int i = 0; i < LD_; ++i) {
      bool ok = (qj > 0) && (dm[i] > 0);
      float v = ok ? (float)sSc[i * 40 + j] : -1.0e9f;
      sSq2dT[j * 136 + i] = (__bf16)(ok ? __expf(v - mx) * inv : 0.f);
    }
  }
  __syncthreads();

  // ---- stage Uq^T (bf16 [N=h][K=j]) from the fp32 LDS stage ----
  for (int idx = tid; idx < H_ * LQ_; idx += 256) {
    int h = idx >> 5, j = idx & 31;
    sUqT[h * 40 + j] = (__bf16)sUqStage[j * H_ + h];
  }
  __syncthreads();

  // ---- P4: T = S_q2d^T @ Ud  [32,256], stored transposed sTT [h][j] ----
  for (int t = 0; t < 4; ++t) {
    int id = wave + 8 * t;              // 32 tiles: 2 mtiles x 16 ntiles
    int mt = id >> 4, nt = id & 15;
    v8f cacc = {};
    for (int kk = 0; kk < LD_; kk += 32) {
      v16bf a  = load_A_lds(sSq2dT, 136, mt * 16, kk);
      v16bf bb = load_B_g_kmajor(Ud, H_, kk, nt * 16);   // B[k=i][n=h] = Ud[i][h]
      cacc = wmma_bf16(a, bb, cacc);
    }
    int n, hx; lane_mh(n, hx);
#pragma unroll
    for (int r = 0; r < 8; ++r)
      sTT[(nt * 16 + n) * 40 + mt * 16 + hx * 8 + r] = (__bf16)cacc[r];
  }
  __syncthreads();

  // ---- process rows in two halves of 64 with online softmax ----
  for (int hf = 0; hf < 2; ++hf) {
    const int rbase = hf * 64;

    // P3h: A_d2q rows = S_d2q @ Uq   (K = 32, single WMMA per tile)
    for (int t = 0; t < 8; ++t) {
      int id = wave + 8 * t;            // 64 tiles: 4 mtiles x 16 ntiles
      int mt = id >> 4, nt = id & 15;
      v16bf a  = load_A_lds(sSd2q, 40, rbase + mt * 16, 0);
      v16bf bb = load_B_lds(sUqT, 40, nt * 16, 0);
      v8f cacc = {};
      cacc = wmma_bf16(a, bb, cacc);
      int n, hx; lane_mh(n, hx);
#pragma unroll
      for (int r = 0; r < 8; ++r)
        sAd2q[(mt * 16 + hx * 8 + r) * 264 + nt * 16 + n] = (__bf16)cacc[r];
    }
    // P5h: A_q2d rows = S_d2q @ T    (K = 32)
    for (int t = 0; t < 8; ++t) {
      int id = wave + 8 * t;
      int mt = id >> 4, nt = id & 15;
      v16bf a  = load_A_lds(sSd2q, 40, rbase + mt * 16, 0);
      v16bf bb = load_B_lds(sTT, 40, nt * 16, 0);
      v8f cacc = {};
      cacc = wmma_bf16(a, bb, cacc);
      int n, hx; lane_mh(n, hx);
#pragma unroll
      for (int r = 0; r < 8; ++r)
        sAq2d[(mt * 16 + hx * 8 + r) * 264 + nt * 16 + n] = (__bf16)cacc[r];
    }
    __syncthreads();
    if (tid < 64) slog[tid] = 0.f;
    __syncthreads();

    // P6h: M = tanh(V_d @ Ws_w^T + b), logits l[i] = M[i]·wcs
    {
      const int mt    = wave & 3;
      const int ntb   = (wave >> 2) * 8;
      const int grow0 = rbase + mt * 16;
      v8f cacc[8];
#pragma unroll
      for (int q = 0; q < 8; ++q) cacc[q] = (v8f){};
#pragma unroll 1
      for (int ch = 0; ch < 4; ++ch) {
        for (int kk = 0; kk < H_; kk += 32) {
          v16bf a;
          if (ch == 0)      a = load_A_g(Ud, H_, grow0, kk);
          else if (ch == 1) a = load_A_lds(sAd2q, 264, mt * 16, kk);
          else if (ch == 2) a = load_A_prod(Ud, H_, grow0, sAd2q, 264, mt * 16, kk);
          else              a = load_A_prod(Ud, H_, grow0, sAq2d, 264, mt * 16, kk);
#pragma unroll
          for (int q = 0; q < 8; ++q) {
            v16bf bb = load_B_g(Ws_w, FH_, (ntb + q) * 16, ch * H_ + kk);
            cacc[q] = wmma_bf16(a, bb, cacc[q]);
          }
        }
      }
      int n, hx; lane_mh(n, hx);
#pragma unroll
      for (int q = 0; q < 8; ++q) {
        int col = (ntb + q) * 16 + n;
        float wsb = Ws_b[col];
        float wcs = wcs_w[col];
#pragma unroll
        for (int r = 0; r < 8; ++r) {
          float v = tanhf(cacc[q][r] + wsb) * wcs;
          v += __shfl_xor(v, 1, 32);
          v += __shfl_xor(v, 2, 32);
          v += __shfl_xor(v, 4, 32);
          v += __shfl_xor(v, 8, 32);
          if (n == 0) atomicAdd(&slog[mt * 16 + hx * 8 + r], v);
        }
      }
    }
    __syncthreads();

    // online softmax bookkeeping for this half
    if (tid == 0) {
      const float wb = wcs_b[0];
      float mold = stats[0];
      float mloc = -3.0e38f;
      for (int i = 0; i < 64; ++i) {
        if (dm[rbase + i] > 0) {
          float l = slog[i] + wb;
          slog[i] = l;
          mloc = fmaxf(mloc, l);
        }
      }
      float mnew = fmaxf(mold, mloc);
      float scale = (mnew > -1.0e38f) ? __expf(mold - mnew) : 1.f;
      float srun = stats[1] * scale;
      for (int i = 0; i < 64; ++i) {
        float e = (dm[rbase + i] > 0 && mnew > -1.0e38f)
                      ? __expf(slog[i] - mnew) : 0.f;
        slog[i] = e;
        srun += e;
      }
      stats[0] = mnew; stats[1] = srun; stats[2] = scale;
    }
    __syncthreads();

    // acc[c] = acc[c]*scale + sum_i e_i * V_d[i,c]
    {
      const float scale = stats[2];
      for (int c0 = tid; c0 < FH_; c0 += 256) {
        float v = acc[c0] * scale;
        const int ch = c0 >> 8, cc = c0 & 255;
        for (int i = 0; i < 64; ++i) {
          float e = slog[i];
          if (e != 0.f) {
            float vd;
            if (ch == 0)      vd = Ud[(size_t)(rbase + i) * H_ + cc];
            else if (ch == 1) vd = (float)sAd2q[i * 264 + cc];
            else if (ch == 2) vd = Ud[(size_t)(rbase + i) * H_ + cc] * (float)sAd2q[i * 264 + cc];
            else              vd = Ud[(size_t)(rbase + i) * H_ + cc] * (float)sAq2d[i * 264 + cc];
            v += e * vd;
          }
        }
        acc[c0] = v;
      }
    }
    __syncthreads();
  }

  // ---- alpha = acc / sum ----
  {
    const float s = stats[1];
    const float inv = (s > 0.f) ? 1.f / s : 0.f;
    for (int c0 = tid; c0 < FH_; c0 += 256)
      alpha_out[(size_t)bs * FH_ + c0] = acc[c0] * inv;
  }
}

// ---------------------------------------------------------------------------
// Kernel 2: outer attention over S -> y [16][1024]
// ---------------------------------------------------------------------------
__global__ __launch_bounds__(256)
void dhia_outer_kernel(const float* __restrict__ alpha,
                       const int* __restrict__ sent_mask,
                       const float* __restrict__ Wd_w, const float* __restrict__ Wd_b,
                       const float* __restrict__ wcd_w, const float* __restrict__ wcd_b,
                       float* __restrict__ out) {
  __shared__ float slog[32];
  __shared__ float sp[32];
  const int tid  = threadIdx.x;
  const int wave = tid >> 5;
  const int b    = blockIdx.x;
  const float* al = alpha + (size_t)b * S_ * FH_;

  if (tid < 32) slog[tid] = 0.f;
  __syncthreads();

  // G = tanh(alpha @ Wd_w^T + b); logits over S via wcd
  for (int t = 0; t < 4; ++t) {
    int id = wave + 8 * t;              // 32 tiles: 2 mtiles x 16 ntiles
    int mt = id >> 4, nt = id & 15;
    v8f cacc = {};
    for (int kk = 0; kk < FH_; kk += 32) {
      v16bf a  = load_A_g(al, FH_, mt * 16, kk);
      v16bf bb = load_B_g(Wd_w, FH_, nt * 16, kk);
      cacc = wmma_bf16(a, bb, cacc);
    }
    int n, hx; lane_mh(n, hx);
    int col = nt * 16 + n;
    float wb = Wd_b[col];
    float wc = wcd_w[col];
#pragma unroll
    for (int r = 0; r < 8; ++r) {
      float v = tanhf(cacc[r] + wb) * wc;
      v += __shfl_xor(v, 1, 32);
      v += __shfl_xor(v, 2, 32);
      v += __shfl_xor(v, 4, 32);
      v += __shfl_xor(v, 8, 32);
      if (n == 0) atomicAdd(&slog[mt * 16 + hx * 8 + r], v);
    }
  }
  __syncthreads();

  if (tid == 0) {
    const int* sm = sent_mask + b * S_;
    const float wb = wcd_b[0];
    float mx = -3.0e38f;
    for (int j = 0; j < S_; ++j)
      if (sm[j] > 0) mx = fmaxf(mx, slog[j] + wb);
    float ssum = 0.f;
    for (int j = 0; j < S_; ++j) {
      float e = (sm[j] > 0 && mx > -1.0e38f) ? __expf(slog[j] + wb - mx) : 0.f;
      sp[j] = e;
      ssum += e;
    }
    float inv = (ssum > 0.f) ? 1.f / ssum : 0.f;
    for (int j = 0; j < S_; ++j) sp[j] *= inv;
  }
  __syncthreads();

  for (int c0 = tid; c0 < FH_; c0 += 256) {
    float v = 0.f;
    for (int j = 0; j < S_; ++j) v += sp[j] * al[(size_t)j * FH_ + c0];
    out[(size_t)b * FH_ + c0] = v;
  }
}

// ---------------------------------------------------------------------------
extern "C" void kernel_launch(void* const* d_in, const int* in_sizes, int n_in,
                              void* d_out, int out_size, void* d_ws, size_t ws_size,
                              hipStream_t stream) {
  const float* U_d      = (const float*)d_in[0];
  const float* U_q      = (const float*)d_in[1];
  const int*   d_mask   = (const int*)d_in[2];
  const int*   q_mask   = (const int*)d_in[3];
  const int*   sent_mask= (const int*)d_in[4];
  const float* wc_w     = (const float*)d_in[5];
  const float* wc_b     = (const float*)d_in[6];
  const float* Ws_w     = (const float*)d_in[7];
  const float* Ws_b     = (const float*)d_in[8];
  const float* wcs_w    = (const float*)d_in[9];
  const float* wcs_b    = (const float*)d_in[10];
  const float* Wd_w     = (const float*)d_in[11];
  const float* Wd_b     = (const float*)d_in[12];
  const float* wcd_w    = (const float*)d_in[13];
  const float* wcd_b    = (const float*)d_in[14];

  float* alpha = (float*)d_ws;          // [512][1024] f32 = 2 MB scratch
  float* out   = (float*)d_out;         // [16][1024] f32

  (void)in_sizes; (void)n_in; (void)out_size; (void)ws_size;

  hipFuncSetAttribute(reinterpret_cast<const void*>(dhia_inner_kernel),
                      hipFuncAttributeMaxDynamicSharedMemorySize, SMEM_BYTES);

  dhia_inner_kernel<<<B_ * S_, 256, SMEM_BYTES, stream>>>(
      U_d, U_q, d_mask, q_mask, wc_w, wc_b, Ws_w, Ws_b, wcs_w, wcs_b, alpha);

  dhia_outer_kernel<<<B_, 256, 0, stream>>>(
      alpha, sent_mask, Wd_w, Wd_b, wcd_w, wcd_b, out);
}
